// GCN3_69672959475883
// MI455X (gfx1250) — matrix operand order repaired and checked by reading, not compiled, not run on hardware
//
#include <hip/hip_runtime.h>
#include <hip/hip_bf16.h>

// ---------------------------------------------------------------------------
// GCN 3-layer forward for MI455X (gfx1250, wave32, WMMA).
// Heavy op: x[ N x N ] @ W1[ N x 64 ]  (N=8192) -> v_wmma_f32_16x16x32_bf16.
// Double-buffered LDS pipeline. Weights are pre-transposed to column-major
// bf16 so B tiles stream straight into transposed LDS via CDNA5 async
// global->LDS copies (GLOBAL_LOAD_ASYNC_TO_LDS_B128, ASYNCcnt); fragments are
// pure ds_load_b128. A tiles are converted f32->bf16 through registers.
// ---------------------------------------------------------------------------

typedef __attribute__((ext_vector_type(16))) __bf16 v16bf;
typedef __attribute__((ext_vector_type(8)))  __bf16 v8bf;
typedef __attribute__((ext_vector_type(8)))  float  v8f;

__device__ __forceinline__ __bf16 to_bf(float f) {
  union { float f; unsigned u; } a; a.f = f;
  unsigned r = a.u + 0x7FFFu + ((a.u >> 16) & 1u);   // round-to-nearest-even
  union { unsigned short s; __bf16 b; } o; o.s = (unsigned short)(r >> 16);
  return o.b;
}
__device__ __forceinline__ __bf16 to_bf(__bf16 b) { return b; }

// ---------------------------------------------------------------------------
// C[M x 64] = A[M x K] @ B^T, where Bt is [64 x K] column-major bf16 weights.
// Block tile: 32 rows x 64 cols, 8 waves, each wave one 16x16 WMMA tile
// (2 x 4 arrangement). K-step = 32, double-buffered LDS.
// ---------------------------------------------------------------------------
template <typename TA>
__global__ __launch_bounds__(256) void gemm_n64_wmma(const TA* __restrict__ A,
                                                     const __bf16* __restrict__ Bt,
                                                     float* __restrict__ C,
                                                     int K) {
  // Row pitch 48 half-words = 96B: multiple of 16 -> aligned b128 LDS ops
  __shared__ __bf16 lA[2][32][48];     // [buf][m][k]  A tile, bf16
  __shared__ __bf16 lBt[2][64][48];    // [buf][n][k]  B tile (transposed)

  const int tid  = threadIdx.x;
  const int lane = tid & 31;
  const int wid  = tid >> 5;            // 0..7
  const int m0w  = (wid >> 2) * 16;     // wave tile row offset within block
  const int n0w  = (wid & 3) * 16;      // wave tile col offset within block
  const int m0   = blockIdx.x * 32;     // block row offset
  const int half = lane >> 4;           // 0 / 1
  const int r16  = lane & 15;

  // staging coordinates
  const int ar = tid >> 3;              // 0..31  A row,  one float4 per thread
  const int ac = (tid & 7) * 4;         // 0..28  A col
  const int bn = tid >> 2;              // 0..63  B column (n)
  const int bk = (tid & 3) * 8;         // 0..24  B k-chunk (8 bf16 = 16B)

  const TA*     Ap  = A  + (size_t)(m0 + ar) * K + ac;
  const __bf16* Btp = Bt + (size_t)bn * K + bk;

  v8f acc = {};
  const int nk = K >> 5;                // K-steps of 32

  float a_f[4];

  auto load_A = [&](int kt) {
    const size_t ko = (size_t)kt * 32;
    if constexpr (sizeof(TA) == 4) {
      float4 t = *(const float4*)(Ap + ko);
      a_f[0] = t.x; a_f[1] = t.y; a_f[2] = t.z; a_f[3] = t.w;
    } else {
      const __bf16* p = (const __bf16*)(Ap + ko);
      a_f[0] = (float)p[0]; a_f[1] = (float)p[1];
      a_f[2] = (float)p[2]; a_f[3] = (float)p[3];
    }
  };

  auto store_A = [&](int buf) {
    union { __bf16 h[4]; uint2 u; } pa;
    pa.h[0] = to_bf(a_f[0]); pa.h[1] = to_bf(a_f[1]);
    pa.h[2] = to_bf(a_f[2]); pa.h[3] = to_bf(a_f[3]);
    *(uint2*)&lA[buf][ar][ac] = pa.u;       // one ds_store_b64
  };

  // CDNA5 async copy: 16B of column-major weights straight into transposed
  // LDS tile -- no VGPR staging, tracked by ASYNCcnt.
  auto async_B = [&](int kt, int buf) {
    const __bf16* src = Btp + (size_t)kt * 32;                // 16B aligned
    const unsigned dst = (unsigned)(size_t)&lBt[buf][bn][bk]; // LDS byte offset
    asm volatile("global_load_async_to_lds_b128 %0, %1, off"
                 :: "v"(dst), "v"(src) : "memory");
  };

  // ---- prologue: tile 0 into buffer 0 -------------------------------------
  load_A(0);
  async_B(0, 0);
  store_A(0);
  asm volatile("s_wait_asynccnt 0x0" ::: "memory");
  __syncthreads();

  for (int kt = 0; kt < nk; ++kt) {
    const int  cur  = kt & 1;
    const bool more = (kt + 1) < nk;

    // issue next tile's loads early (latency overlapped with WMMA below)
    if (more) {
      load_A(kt + 1);
      async_B(kt + 1, cur ^ 1);
      // L2 prefetch two tiles ahead (global_prefetch_b8)
      if (kt + 2 < nk)
        __builtin_prefetch(Ap + (size_t)(kt + 2) * 32, 0, 1);
    }

    // ---- A fragment (ISA 16-bit A 16x32 layout) ---------------------------
    // lane<16: row=r16, elems 0..7 = K 0..7,  elems 8..15 = K 16..23
    // lane>=16: row=r16, elems 0..7 = K 8..15, elems 8..15 = K 24..31
    v8bf alo = *(const v8bf*)&lA[cur][m0w + r16][half * 8];
    v8bf ahi = *(const v8bf*)&lA[cur][m0w + r16][half * 8 + 16];
    v16bf a = __builtin_shufflevector(alo, ahi, 0,1,2,3,4,5,6,7,8,9,10,11,12,13,14,15);

    // ---- B fragment: lane holds column n0w+r16, K = half*8..+7, 16+half*8..+7 --
    v8bf blo = *(const v8bf*)&lBt[cur][n0w + r16][half * 8];
    v8bf bhi = *(const v8bf*)&lBt[cur][n0w + r16][half * 8 + 16];
    v16bf b = __builtin_shufflevector(blo, bhi, 0,1,2,3,4,5,6,7,8,9,10,11,12,13,14,15);

    acc = __builtin_amdgcn_wmma_f32_16x16x32_bf16(false, a, false, b,
                                                  (short)0, acc, false, false);

    // stage next A tile into the other buffer (s_wait_loadcnt lands here)
    if (more) store_A(cur ^ 1);
    asm volatile("s_wait_asynccnt 0x0" ::: "memory");
    __syncthreads();
  }

  // ---- store: C/D layout -> VGPR v holds row half*8+v, col = r16 ----------
  #pragma unroll
  for (int v = 0; v < 8; ++v) {
    const int row = m0 + m0w + half * 8 + v;
    C[(size_t)row * 64 + (n0w + r16)] = acc[v];
  }
}

// ---------------------------------------------------------------------------
// Edge-parallel SpMM scatter: T[row[e], :] += val[e] * S[col[e], :]
// ---------------------------------------------------------------------------
template <int F>
__global__ __launch_bounds__(256) void spmm_scatter(const int* __restrict__ row,
                                                    const int* __restrict__ col,
                                                    const float* __restrict__ val,
                                                    const float* __restrict__ S,
                                                    float* __restrict__ T, int E) {
  constexpr int LF = (F == 64) ? 6 : 3;
  const long long gid = (long long)blockIdx.x * blockDim.x + threadIdx.x;
  const int e = (int)(gid >> LF);
  const int f = (int)(gid & (F - 1));
  if (e < E) {
    const float v = val[e];
    const int   c = col[e];
    const int   r = row[e];
    atomicAdd(&T[(size_t)r * F + f], v * S[(size_t)c * F + f]);
  }
}

// out = relu(T + b) stored as bf16 for the next WMMA GEMM (F = 64 fixed)
__global__ __launch_bounds__(256) void bias_relu_bf16(const float* __restrict__ T,
                                                      const float* __restrict__ b,
                                                      __bf16* __restrict__ H, int total) {
  const int i = blockIdx.x * 256 + threadIdx.x;
  if (i < total) {
    const float x = T[i] + b[i & 63];
    H[i] = to_bf(x > 0.f ? x : 0.f);
  }
}

// S3[N x 8] = H2[N x 64](bf16) @ W3[64 x 8](f32) -- tiny, plain VALU
__global__ __launch_bounds__(256) void gemm_h2w3(const __bf16* __restrict__ H2,
                                                 const float* __restrict__ W3,
                                                 float* __restrict__ S3, int N) {
  const int gid = blockIdx.x * 256 + threadIdx.x;
  const int r = gid >> 3, c = gid & 7;
  if (r < N) {
    float s = 0.f;
    #pragma unroll
    for (int k = 0; k < 64; ++k) s += (float)H2[(size_t)r * 64 + k] * W3[k * 8 + c];
    S3[gid] = s;
  }
}

// per-row: v = T3[row]+b3 ; h = log_softmax(v) ; out[c] += h[c] * Wlin[row]
__global__ __launch_bounds__(256) void lsm_dot(const float* __restrict__ T3,
                                               const float* __restrict__ b3,
                                               const float* __restrict__ Wlin,
                                               float* __restrict__ out, int N) {
  __shared__ float sred[8];
  if (threadIdx.x < 8) sred[threadIdx.x] = 0.f;
  __syncthreads();
  const int r = blockIdx.x * 256 + threadIdx.x;
  if (r < N) {
    float v[8], m = -3.402823e38f;
    #pragma unroll
    for (int c = 0; c < 8; ++c) { v[c] = T3[(size_t)r * 8 + c] + b3[c]; m = fmaxf(m, v[c]); }
    float s = 0.f;
    #pragma unroll
    for (int c = 0; c < 8; ++c) s += __expf(v[c] - m);
    const float lse = __logf(s) + m;
    const float w = Wlin[r];
    #pragma unroll
    for (int c = 0; c < 8; ++c) atomicAdd(&sred[c], (v[c] - lse) * w);
  }
  __syncthreads();
  if (threadIdx.x < 8) atomicAdd(&out[threadIdx.x], sred[threadIdx.x]);
}

__global__ void zero_f32(float* p, int n) {
  const int i = blockIdx.x * 256 + threadIdx.x;
  if (i < n) p[i] = 0.f;
}
__global__ void init_out(float* out, const float* blin) {
  if (threadIdx.x < 8) out[threadIdx.x] = blin[0];
}
// W [K x 64] f32 row-major  ->  Wt [64 x K] bf16 (column-major weights)
__global__ void cvt_bf16_T(const float* __restrict__ src, __bf16* __restrict__ dst,
                           int K, int total) {
  const int i = blockIdx.x * 256 + threadIdx.x;
  if (i < total) {
    const int k = i >> 6, n = i & 63;
    dst[(size_t)n * K + k] = to_bf(src[i]);
  }
}

// ---------------------------------------------------------------------------
extern "C" void kernel_launch(void* const* d_in, const int* in_sizes, int n_in,
                              void* d_out, int out_size, void* d_ws, size_t ws_size,
                              hipStream_t stream) {
  const float* x    = (const float*)d_in[0];
  const int*   arow = (const int*)  d_in[1];
  const int*   acol = (const int*)  d_in[2];
  const float* aval = (const float*)d_in[3];
  const float* W1   = (const float*)d_in[4];
  const float* b1   = (const float*)d_in[5];
  const float* W2   = (const float*)d_in[6];
  const float* b2   = (const float*)d_in[7];
  const float* W3   = (const float*)d_in[8];
  const float* b3   = (const float*)d_in[9];
  const float* Wlin = (const float*)d_in[10];
  const float* blin = (const float*)d_in[11];
  float* out = (float*)d_out;

  const int N = in_sizes[10];   // 8192 nodes == nfeat
  const int E = in_sizes[1];    // edges

  // ---- workspace carve-out (256B aligned) ---------------------------------
  char* ws = (char*)d_ws;
  size_t o = 0;
  auto carve = [&](size_t bytes) -> void* {
    o = (o + 255) & ~(size_t)255;
    void* p = ws + o;
    o += bytes;
    return p;
  };
  float*  T1  = (float*) carve((size_t)N * 64 * 4);
  float*  T2  = (float*) carve((size_t)N * 64 * 4);
  float*  T3  = (float*) carve((size_t)N * 8  * 4);
  float*  S1  = (float*) carve((size_t)N * 64 * 4);
  float*  S2  = (float*) carve((size_t)N * 64 * 4);
  float*  S3  = (float*) carve((size_t)N * 8  * 4);
  __bf16* W1t = (__bf16*)carve((size_t)N * 64 * 2);   // [64 x N] col-major
  __bf16* W2t = (__bf16*)carve((size_t)64 * 64 * 2);  // [64 x 64] col-major
  __bf16* H1  = (__bf16*)carve((size_t)N * 64 * 2);
  __bf16* H2  = (__bf16*)carve((size_t)N * 64 * 2);

  const unsigned nf  = (unsigned)(N * 64);
  const unsigned nc  = (unsigned)(N * 8);
  const unsigned e64 = (unsigned)(((long long)E * 64 + 255) / 256);
  const unsigned e8  = (unsigned)(((long long)E * 8  + 255) / 256);

  // ---- init ---------------------------------------------------------------
  zero_f32<<<(nf + 255) / 256, 256, 0, stream>>>(T1, nf);
  zero_f32<<<(nf + 255) / 256, 256, 0, stream>>>(T2, nf);
  zero_f32<<<(nc + 255) / 256, 256, 0, stream>>>(T3, nc);
  init_out<<<1, 32, 0, stream>>>(out, blin);
  cvt_bf16_T<<<(nf + 255) / 256, 256, 0, stream>>>(W1, W1t, N, nf);
  cvt_bf16_T<<<(64 * 64 + 255) / 256, 256, 0, stream>>>(W2, W2t, 64, 64 * 64);

  // ---- layer 1: S1 = x @ W1 (WMMA), T1 = A_sp @ S1, H1 = relu(T1+b1) ------
  gemm_n64_wmma<float><<<N / 32, 256, 0, stream>>>(x, W1t, S1, N);
  spmm_scatter<64><<<e64, 256, 0, stream>>>(arow, acol, aval, S1, T1, E);
  bias_relu_bf16<<<(nf + 255) / 256, 256, 0, stream>>>(T1, b1, H1, nf);

  // ---- layer 2: S2 = H1 @ W2 (WMMA, K=64) ---------------------------------
  gemm_n64_wmma<__bf16><<<N / 32, 256, 0, stream>>>(H1, W2t, S2, 64);
  spmm_scatter<64><<<e64, 256, 0, stream>>>(arow, acol, aval, S2, T2, E);
  bias_relu_bf16<<<(nf + 255) / 256, 256, 0, stream>>>(T2, b2, H2, nf);

  // ---- layer 3: S3 = H2 @ W3, T3 = A_sp @ S3 ------------------------------
  gemm_h2w3<<<(nc + 255) / 256, 256, 0, stream>>>(H2, W3, S3, N);
  spmm_scatter<8><<<e8, 256, 0, stream>>>(arow, acol, aval, S3, T3, E);

  // ---- bias + log_softmax + (h^T @ Wlin^T + blin) -------------------------
  lsm_dot<<<(N + 255) / 256, 256, 0, stream>>>(T3, b3, Wlin, out, N);
}
